// BlockSparseAttention_63075889709244
// MI455X (gfx1250) — matrix operand order repaired and checked by reading, not compiled, not run on hardware
//
#include <hip/hip_runtime.h>
#include <hip/hip_bf16.h>

typedef __attribute__((ext_vector_type(16))) _Float16 v16h;
typedef __attribute__((ext_vector_type(8)))  _Float16 v8h;
typedef __attribute__((ext_vector_type(8)))  float    v8f;

#define B_   2
#define H_   16
#define S_   2048
#define D_   128
#define BS_  128
#define NB_  16
#define KPAD 136                     // padded LDS row stride in halves (272B -> conflict-free)
#define SCALE_ 0.08838834764831845f  // 1/sqrt(128)

union U32H2 { _Float16 h[2]; unsigned u; };
union FragU { _Float16 h[16]; unsigned u[8]; v16h v; };

// A-matrix f16 fragment (16x32 tile): lane half hf holds cols {16hf..16hf+15}
// of its row -> two contiguous 16B LDS loads.
static __device__ inline v16h load_a_frag(const _Float16* p) {
  v8h lo = *(const v8h*)(p);
  v8h hi = *(const v8h*)(p + 8);
  return __builtin_shufflevector(lo, hi, 0,1,2,3,4,5,6,7,8,9,10,11,12,13,14,15);
}

// B-matrix f16 fragment: lane half hf holds 16 contiguous K rows (cols of Q row) -> 16 f32 cvt.
static __device__ inline v16h cvt_b_frag_f32(const float* p) {
  FragU f;
#pragma unroll
  for (int j = 0; j < 16; ++j) f.h[j] = (_Float16)p[j];
  return f.v;
}

__global__ __launch_bounds__(256)
void BlockSparseAttn_kernel(const float* __restrict__ Q, const float* __restrict__ K,
                            const float* __restrict__ V,
                            const unsigned char* __restrict__ mask,  // JAX bool = 1 byte
                            float* __restrict__ Out) {
  extern __shared__ _Float16 smem[];
  _Float16* sK  = smem;              // 128 x KPAD f16 (K block, row-major)
  _Float16* sVt = smem + BS_ * KPAD; // 128 x KPAD f16 (V block, transposed: [d][k])

  const int ib = blockIdx.x, hd = blockIdx.y, bb = blockIdx.z;
  const int tid  = threadIdx.x;
  const int wave = tid >> 5, lane = tid & 31;
  const int ln = lane & 15, hf = lane >> 4;

  const size_t bh   = ((size_t)bb * H_ + hd) * S_;
  const float* qblk = Q   + (bh + (size_t)ib * BS_) * D_;
  float*       oblk = Out + (bh + (size_t)ib * BS_) * D_;

  // active-j bitmask (diagonal is always set by the reference)
  unsigned abits = 0;
#pragma unroll
  for (int j = 0; j < NB_; ++j) abits |= mask[ib * NB_ + j] ? (1u << j) : 0u;

  // Q^T B-fragments, resident for the whole kernel (row = this wave's query, per lane)
  const float* qrow = qblk + (wave * 16 + ln) * D_;
  v16h qf[4];
#pragma unroll
  for (int c = 0; c < 4; ++c) qf[c] = cvt_b_frag_f32(qrow + 32 * c + 16 * hf);

  v8f zf = {};
  v8f oacc[8];                        // O^T accumulator: tile m -> d in [16m,16m+16), n = q
#pragma unroll
  for (int m = 0; m < 8; ++m) oacc[m] = zf;

  for (unsigned bits = abits; bits; ) {
    const int j = __builtin_ctz(bits);
    bits &= bits - 1u;
    const float* kblk = K + (bh + (size_t)j * BS_) * D_;
    const float* vblk = V + (bh + (size_t)j * BS_) * D_;

    // ---- stage K (f16 row-major) and V^T (f16) into LDS (256 threads, coalesced reads)
#pragma unroll
    for (int it = 0; it < 16; ++it) {
      const int f4  = it * 256 + tid;       // 4096 float4 per 128x128 block
      const int row = f4 >> 5;
      const int c4  = (f4 & 31) * 4;
      float4 kv = ((const float4*)kblk)[f4];
      U32H2 a, b;
      a.h[0] = (_Float16)kv.x; a.h[1] = (_Float16)kv.y;
      b.h[0] = (_Float16)kv.z; b.h[1] = (_Float16)kv.w;
      uint2 p2; p2.x = a.u; p2.y = b.u;
      *(uint2*)&sK[row * KPAD + c4] = p2;   // ds_store_b64
      float4 vv = ((const float4*)vblk)[f4];
      sVt[(c4 + 0) * KPAD + row] = (_Float16)vv.x;   // scattered transpose stores
      sVt[(c4 + 1) * KPAD + row] = (_Float16)vv.y;
      sVt[(c4 + 2) * KPAD + row] = (_Float16)vv.z;
      sVt[(c4 + 3) * KPAD + row] = (_Float16)vv.w;
    }
    __syncthreads();

    // ---- prefetch next active block while computing this one (global_prefetch_b8)
    if (bits) {
      const int jn = __builtin_ctz(bits);
      __builtin_prefetch(K + (bh + (size_t)jn * BS_) * D_ + tid * 64, 0, 0);
      __builtin_prefetch(V + (bh + (size_t)jn * BS_) * D_ + tid * 64, 0, 0);
    }

    // ---- GEMM1: S^T = K * Q^T   (M=128 keys, N=16 queries, K=128 features)
    v8f sacc[8];
#pragma unroll
    for (int m = 0; m < 8; ++m) {
      v8f c = zf;
#pragma unroll
      for (int kc = 0; kc < 4; ++kc) {
        v16h a = load_a_frag(&sK[(16 * m + ln) * KPAD + 32 * kc + 16 * hf]);
        c = __builtin_amdgcn_wmma_f32_16x16x32_f16(false, a, false, qf[kc],
                                                   (short)0, c, false, false);
      }
      sacc[m] = c;
    }

    // ---- per-block softmax over k: 64 local values + partner lane (shfl_xor 16)
    float mx = -3.0e38f;
#pragma unroll
    for (int m = 0; m < 8; ++m)
#pragma unroll
      for (int e = 0; e < 8; ++e) mx = fmaxf(mx, sacc[m][e]);
    mx = fmaxf(mx, __shfl_xor(mx, 16, 32));
    float sum = 0.0f;
#pragma unroll
    for (int m = 0; m < 8; ++m)
#pragma unroll
      for (int e = 0; e < 8; ++e) {
        float p = __expf((sacc[m][e] - mx) * SCALE_);
        sacc[m][e] = p;
        sum += p;
      }
    sum += __shfl_xor(sum, 16, 32);
    const float rs = 1.0f / sum;

    // ---- P^T C-layout -> f16 B-fragments: pack pairs + one xor-16 exchange
    unsigned pk[8][4], px[8][4];
#pragma unroll
    for (int m = 0; m < 8; ++m)
#pragma unroll
      for (int w = 0; w < 4; ++w) {
        U32H2 u;
        u.h[0] = (_Float16)(sacc[m][2 * w]     * rs);
        u.h[1] = (_Float16)(sacc[m][2 * w + 1] * rs);
        pk[m][w] = u.u;
        px[m][w] = __shfl_xor(u.u, 16, 32);
      }
    v16h pf[4];
#pragma unroll
    for (int c = 0; c < 4; ++c) {
      FragU f;
#pragma unroll
      for (int w = 0; w < 4; ++w) {
        f.u[w]     = hf ? px[2 * c + 1][w] : pk[2 * c][w];      // K rows 16hf..+7
        f.u[4 + w] = hf ? pk[2 * c + 1][w] : px[2 * c][w];      // K rows 16hf+8..+15
      }
      pf[c] = f.v;
    }

    // ---- GEMM2: O^T += V^T * P^T   (M=128 features, N=16 queries, K=128 keys)
#pragma unroll
    for (int m = 0; m < 8; ++m) {
      v8f c = oacc[m];
#pragma unroll
      for (int kc = 0; kc < 4; ++kc) {
        v16h a = load_a_frag(&sVt[(16 * m + ln) * KPAD + 32 * kc + 16 * hf]);
        c = __builtin_amdgcn_wmma_f32_16x16x32_f16(false, a, false, pf[kc],
                                                   (short)0, c, false, false);
      }
      oacc[m] = c;
    }
    __syncthreads();   // all waves done with LDS before restaging
  }

  // ---- store: lane (q=ln, hf) holds O[q][16m + 8hf + v] -> 16B stores
  float* orow = oblk + (wave * 16 + ln) * D_;
#pragma unroll
  for (int m = 0; m < 8; ++m) {
    const int d0 = 16 * m + 8 * hf;
    float4 lo = { oacc[m][0], oacc[m][1], oacc[m][2], oacc[m][3] };
    float4 hi = { oacc[m][4], oacc[m][5], oacc[m][6], oacc[m][7] };
    *(float4*)(orow + d0)     = lo;
    *(float4*)(orow + d0 + 4) = hi;
  }
}

extern "C" void kernel_launch(void* const* d_in, const int* in_sizes, int n_in,
                              void* d_out, int out_size, void* d_ws, size_t ws_size,
                              hipStream_t stream) {
  (void)in_sizes; (void)n_in; (void)out_size; (void)d_ws; (void)ws_size;
  const float* q = (const float*)d_in[0];
  const float* k = (const float*)d_in[1];
  const float* v = (const float*)d_in[2];
  const unsigned char* mask = (const unsigned char*)d_in[3];  // jnp bool_ = 1 byte/elem
  // d_in[4] = block_size scalar (128), compiled in as a constant
  float* out = (float*)d_out;

  dim3 grid(NB_, H_, B_);           // one 128x128 query block per workgroup
  dim3 block(256);                  // 8 wave32 waves
  size_t lds_bytes = 2u * BS_ * KPAD * sizeof(_Float16);  // 69632 B (K + V^T in f16)
  BlockSparseAttn_kernel<<<grid, block, lds_bytes, stream>>>(q, k, v, mask, out);
}
